// ESRNN_10917806867105
// MI455X (gfx1250) — matrix operand order repaired and checked by reading, not compiled, not run on hardware
//
#include <hip/hip_runtime.h>
#include <hip/hip_bf16.h>

// ---------------------------------------------------------------------------
// ESRNN forward for MI455X (gfx1250, wave32, WMMA).
//   1. es_forward_kernel      : tiny sequential smoothing scan (1 block).
//   2. build_windows_kernel   : windows_x (bf16, padded to 32) + windows_y.
//   3. pack_b_kernel          : pack [Wh;Wx] / mlp_w / ada_w into WMMA
//                               B-fragment-major bf16 (L2-resident).
//   4. lstm_scan_kernel       : persistent kernel, grid barrier per time step,
//                               fused g = [h|x] @ [Wh;Wx] + b via
//                               v_wmma_f32_16x16x32_bf16, gates on accum.
//   5. residual / MLP(tanh) / adapter GEMMs via the same WMMA path
//      (4-wide N-tile blocking for A-fragment reuse).
// ---------------------------------------------------------------------------

typedef __attribute__((ext_vector_type(16))) __bf16 v16bf;
typedef __attribute__((ext_vector_type(8)))  float  v8f;

#define DEV __device__ __forceinline__

DEV unsigned short f2bf(float f) {
  unsigned u = __builtin_bit_cast(unsigned, f);
  u += 0x7FFFu + ((u >> 16) & 1u);          // round-to-nearest-even
  return (unsigned short)(u >> 16);
}
DEV float bf2f(unsigned short h) {
  unsigned u = ((unsigned)h) << 16;
  return __builtin_bit_cast(float, u);
}
DEV float sigm(float x) { return 1.0f / (1.0f + __expf(-x)); }

// A-fragment (16x32 bf16) from row-major [rows x stride] at column kcol.
// ISA layout: lane m = lane&15; dword p holds K pair:
//   K = (p>>2)*16 + (lane>>4)*8 + (p&3)*2
// K pairs are memory-contiguous in two 16B runs -> compiler emits 2x b128.
DEV v16bf load_a_frag(const unsigned short* base, int stride, int kcol, int lane) {
  union { unsigned u32[8]; v16bf v; } t;
  const unsigned short* rp = base + (size_t)(lane & 15) * stride + kcol;
  int kh = (lane >> 4) << 3;
#pragma unroll
  for (int p = 0; p < 8; ++p) {
    int K = ((p >> 2) << 4) + kh + ((p & 3) << 1);
    t.u32[p] = *(const unsigned*)(rp + K);
  }
  return t.v;
}

// B-fragment (32x16 bf16) from fragment-major packed weights:
// pack[((kc*NT + nt)*32 + lane)*16 + e], lane: n = lane&15, K = (lane>>4)*8 + (e&7) + (e>>3)*16
DEV v16bf load_b_frag(const unsigned short* pack, int kc, int nt, int NT, int lane) {
  const unsigned short* bp = pack + ((((size_t)kc * NT + nt) * 32) + lane) * 16;
  return *(const v16bf*)bp;   // 32B per lane, 32B aligned
}

DEV v8f wmma_bf16(v16bf a, v16bf b, v8f c) {
  return __builtin_amdgcn_wmma_f32_16x16x32_bf16(
      /*neg_a=*/false, a, /*neg_b=*/false, b,
      /*c_mod=*/(short)0, c, /*reuse_a=*/false, /*reuse_b=*/false);
}

// ---------------------------------------------------------------------------
// 1. Exponential smoothing forward (64 series, sequential over 1071 steps)
// ---------------------------------------------------------------------------
__global__ void es_forward_kernel(const float* __restrict__ y,
                                  const int* __restrict__ idxs,
                                  const float* __restrict__ lev_sms,
                                  const float* __restrict__ seas_sms,
                                  const float* __restrict__ init_seas,
                                  float* __restrict__ levels,   // (64,1071) -> d_out
                                  float* __restrict__ seas) {   // (64,1095) -> ws
  int s = threadIdx.x;
  if (s >= 64) return;
  __shared__ float q[64][24];
  int id = idxs[s];
  float lev = sigm(lev_sms[id]);
  float sea = sigm(seas_sms[id]);
  float is0 = 0.f;
  for (int j = 0; j < 24; ++j) {
    float v = __expf(init_seas[id * 24 + j]);
    if (j == 0) is0 = v;
    seas[s * 1095 + j] = v;
    q[s][(j + 23) % 24] = v;   // q0 = [iseas[1:], iseas[0]]
  }
  seas[s * 1095 + 24] = is0;
  float level = y[s * 1071 + 0] / is0;
  levels[s * 1071 + 0] = level;
  int head = 0;
  for (int t = 1; t < 1071; ++t) {
    float yt = y[s * 1071 + t];
    float st = q[s][head];
    float nl = lev * (yt / st) + (1.f - lev) * level;
    float ns = sea * (yt / nl) + (1.f - sea) * st;
    q[s][head] = ns;
    head = head + 1; if (head == 24) head = 0;
    level = nl;
    levels[s * 1071 + t] = nl;
    seas[s * 1095 + 24 + t] = ns;
  }
}

// ---------------------------------------------------------------------------
// 2. Window construction: X0 bf16 (W*64, 32) [x(24)|cats(6)|pad(2)], windows_y
// ---------------------------------------------------------------------------
__global__ void build_windows_kernel(const float* __restrict__ y,
                                     const float* __restrict__ categories,
                                     const float* __restrict__ levels,
                                     const float* __restrict__ seas,
                                     unsigned short* __restrict__ X0,
                                     float* __restrict__ windows_y) {
  int idx = blockIdx.x * blockDim.x + threadIdx.x;
  if (idx >= 1024 * 64) return;
  int w = idx >> 6;
  int s = idx & 63;
  float le = levels[s * 1071 + 24 + w];
  unsigned short* xr = X0 + (size_t)idx * 32;
  for (int k = 0; k < 24; ++k)
    xr[k] = f2bf(__logf(y[s * 1071 + w + k] / seas[s * 1095 + w + k] / le));
  for (int k = 0; k < 6; ++k)
    xr[24 + k] = f2bf(categories[s * 6 + k]);
  xr[30] = 0; xr[31] = 0;
  float* wy = windows_y + (size_t)idx * 24;
  for (int j = 0; j < 24; ++j) {
    int tt = 24 + w + j;
    wy[j] = __logf(y[s * 1071 + tt] / seas[s * 1095 + tt]) / le;
  }
}

// ---------------------------------------------------------------------------
// 3. Pack f32 weights [W0(rows0); W1(rows1)] (row-major, ncols) into
//    B-fragment-major bf16. dst element idx == runtime load_b_frag index.
// ---------------------------------------------------------------------------
__global__ void pack_b_kernel(const float* __restrict__ W0, int rows0,
                              const float* __restrict__ W1, int rows1,
                              int ncols, unsigned short* __restrict__ dst,
                              int NT, int total) {
  int idx = blockIdx.x * blockDim.x + threadIdx.x;
  if (idx >= total) return;
  int e    = idx & 15;
  int lane = (idx >> 4) & 31;
  int t1   = idx >> 9;
  int nt   = t1 % NT;
  int kc   = t1 / NT;
  int k = kc * 32 + ((lane >> 4) << 3) + (e & 7) + ((e >> 3) << 4);
  int n = nt * 16 + (lane & 15);
  float v = 0.f;
  if (n < ncols) {
    if (k < rows0)                 v = W0[(size_t)k * ncols + n];
    else if (k - rows0 < rows1)    v = W1[(size_t)(k - rows0) * ncols + n];
  }
  dst[idx] = f2bf(v);
}

// ---------------------------------------------------------------------------
// 4. Persistent dilated-LSTM scan. One grid barrier per time step.
//    g = h_{t-1} @ Wh + x_t @ Wx + b, fused as K=[512(h) | 32*KCx(x)] GEMM.
//    Each wave owns quad (mt, nq): gate tiles nq, nq+32, nq+64, nq+96.
// ---------------------------------------------------------------------------
__global__ void lstm_scan_kernel(const unsigned short* __restrict__ Xin, int xstride, int KCx,
                                 unsigned short* __restrict__ Hseq,
                                 const unsigned short* __restrict__ Wp,
                                 const float* __restrict__ bias,
                                 float* __restrict__ Cst,
                                 unsigned* __restrict__ counter,
                                 int Tsteps, int Br, int nblocks) {
  const int NT = 128;                      // 2048 / 16 output tiles
  int wpb    = blockDim.x >> 5;
  int gwave  = blockIdx.x * wpb + (threadIdx.x >> 5);
  int nwaves = nblocks * wpb;
  int lane   = threadIdx.x & 31;
  int nQuads = (Br >> 4) * 32;

  for (int t = 0; t < Tsteps; ++t) {
    for (int q = gwave; q < nQuads; q += nwaves) {   // uniform per wave
      int mt = q >> 5, nq = q & 31;
      v8f a0 = {}, a1 = {}, a2 = {}, a3 = {};
      if (t > 0) {
        const unsigned short* Hb = Hseq + ((size_t)(t - 1) * Br + mt * 16) * 512;
#pragma unroll 4
        for (int kc = 0; kc < 16; ++kc) {
          v16bf a = load_a_frag(Hb, 512, kc * 32, lane);
          a0 = wmma_bf16(a, load_b_frag(Wp, kc, nq,      NT, lane), a0);
          a1 = wmma_bf16(a, load_b_frag(Wp, kc, nq + 32, NT, lane), a1);
          a2 = wmma_bf16(a, load_b_frag(Wp, kc, nq + 64, NT, lane), a2);
          a3 = wmma_bf16(a, load_b_frag(Wp, kc, nq + 96, NT, lane), a3);
        }
      }
      const unsigned short* Xb = Xin + ((size_t)t * Br + mt * 16) * xstride;
      for (int kc = 0; kc < KCx; ++kc) {
        v16bf a = load_a_frag(Xb, xstride, kc * 32, lane);
        a0 = wmma_bf16(a, load_b_frag(Wp, 16 + kc, nq,      NT, lane), a0);
        a1 = wmma_bf16(a, load_b_frag(Wp, 16 + kc, nq + 32, NT, lane), a1);
        a2 = wmma_bf16(a, load_b_frag(Wp, 16 + kc, nq + 64, NT, lane), a2);
        a3 = wmma_bf16(a, load_b_frag(Wp, 16 + kc, nq + 96, NT, lane), a3);
      }
      int n = nq * 16 + (lane & 15);
      float b0 = bias[n], b1 = bias[512 + n], b2 = bias[1024 + n], b3 = bias[1536 + n];
#pragma unroll
      for (int v = 0; v < 8; ++v) {
        int m = mt * 16 + v + ((lane >> 4) << 3);     // C/D layout
        float gi = a0[v] + b0, gf = a1[v] + b1, gg = a2[v] + b2, go = a3[v] + b3;
        float co = (t > 0) ? Cst[(size_t)m * 512 + n] : 0.f;
        float cn = sigm(gf) * co + sigm(gi) * tanhf(gg);
        float h  = sigm(go) * tanhf(cn);
        Cst[(size_t)m * 512 + n] = cn;
        Hseq[((size_t)t * Br + m) * 512 + n] = f2bf(h);
      }
    }
    // ---- grid-wide barrier (monotonic counter, one signal per block) ----
    __threadfence();
    __syncthreads();
    if (threadIdx.x == 0) {
      atomicAdd(counter, 1u);
      unsigned target = (unsigned)nblocks * (unsigned)(t + 1);
      while (*(volatile unsigned*)counter < target)
        __builtin_amdgcn_s_sleep(1);
    }
    __syncthreads();
    __threadfence();
  }
}

// ---------------------------------------------------------------------------
// 5. Generic WMMA GEMM: out = act(A(bf16, M x 32*KC row-major) @ Bpack + bias)
//    When NT is a multiple of 4, each wave computes 4 consecutive N tiles,
//    reusing the A fragment across 4 WMMAs.
// ---------------------------------------------------------------------------
__global__ void gemm_wmma_kernel(const unsigned short* __restrict__ A, int lda,
                                 const unsigned short* __restrict__ Bp,
                                 const float* __restrict__ bias, int biasN,
                                 int Mtiles, int NT, int KC,
                                 unsigned short* __restrict__ outBf, int ldo,
                                 float* __restrict__ outF, int fcols, int doTanh) {
  int wpb    = blockDim.x >> 5;
  int gwave  = blockIdx.x * wpb + (threadIdx.x >> 5);
  int nwaves = gridDim.x * wpb;
  int lane   = threadIdx.x & 31;

  if ((NT & 3) == 0) {
    // ---- 4-wide N blocking: quad = (mt, nq4), tiles nt0..nt0+3 ----
    int NQ = NT >> 2;
    int total = Mtiles * NQ;
    for (int tile = gwave; tile < total; tile += nwaves) {
      int mt = tile / NQ;
      int nt0 = (tile % NQ) << 2;
      v8f c0 = {}, c1 = {}, c2 = {}, c3 = {};
      const unsigned short* Ab = A + (size_t)mt * 16 * lda;
      for (int kc = 0; kc < KC; ++kc) {
        v16bf a = load_a_frag(Ab, lda, kc * 32, lane);
        c0 = wmma_bf16(a, load_b_frag(Bp, kc, nt0,     NT, lane), c0);
        c1 = wmma_bf16(a, load_b_frag(Bp, kc, nt0 + 1, NT, lane), c1);
        c2 = wmma_bf16(a, load_b_frag(Bp, kc, nt0 + 2, NT, lane), c2);
        c3 = wmma_bf16(a, load_b_frag(Bp, kc, nt0 + 3, NT, lane), c3);
      }
#pragma unroll
      for (int j = 0; j < 4; ++j) {
        v8f acc = (j == 0) ? c0 : (j == 1) ? c1 : (j == 2) ? c2 : c3;
        int n = (nt0 + j) * 16 + (lane & 15);
        float bn = (n < biasN) ? bias[n] : 0.f;
#pragma unroll
        for (int v = 0; v < 8; ++v) {
          int m = mt * 16 + v + ((lane >> 4) << 3);
          float x = acc[v] + bn;
          if (doTanh) x = tanhf(x);
          if (outBf) outBf[(size_t)m * ldo + n] = f2bf(x);
          if (outF && n < fcols) outF[(size_t)m * fcols + n] = x;
        }
      }
    }
  } else {
    // ---- single-tile path (small NT, e.g. adapter NT=2) ----
    int total = Mtiles * NT;
    for (int tile = gwave; tile < total; tile += nwaves) {
      int mt = tile / NT, nt = tile % NT;
      v8f acc = {};
      const unsigned short* Ab = A + (size_t)mt * 16 * lda;
      for (int kc = 0; kc < KC; ++kc) {
        v16bf a = load_a_frag(Ab, lda, kc * 32, lane);
        v16bf b = load_b_frag(Bp, kc, nt, NT, lane);
        acc = wmma_bf16(a, b, acc);
      }
      int n = nt * 16 + (lane & 15);
      float bn = (n < biasN) ? bias[n] : 0.f;
#pragma unroll
      for (int v = 0; v < 8; ++v) {
        int m = mt * 16 + v + ((lane >> 4) << 3);
        float x = acc[v] + bn;
        if (doTanh) x = tanhf(x);
        if (outBf) outBf[(size_t)m * ldo + n] = f2bf(x);
        if (outF && n < fcols) outF[(size_t)m * fcols + n] = x;
      }
    }
  }
}

__global__ void residual_add_kernel(unsigned short* __restrict__ dst,
                                    const unsigned short* __restrict__ add,
                                    size_t n) {
  size_t i = (size_t)blockIdx.x * blockDim.x + threadIdx.x;
  if (i < n) dst[i] = f2bf(bf2f(dst[i]) + bf2f(add[i]));
}

__global__ void init_counters_kernel(unsigned* c) {
  if (threadIdx.x < 64) c[threadIdx.x] = 0u;
}

// ---------------------------------------------------------------------------
// Host orchestration
// ---------------------------------------------------------------------------
extern "C" void kernel_launch(void* const* d_in, const int* in_sizes, int n_in,
                              void* d_out, int out_size, void* d_ws, size_t ws_size,
                              hipStream_t stream) {
  (void)in_sizes; (void)n_in; (void)out_size; (void)ws_size;

  const float* y          = (const float*)d_in[0];
  const int*   idxs       = (const int*)d_in[1];
  const float* categories = (const float*)d_in[2];
  const float* lev_sms    = (const float*)d_in[3];
  const float* seas_sms   = (const float*)d_in[4];
  const float* init_seas  = (const float*)d_in[5];
  const float* Wx[4] = {(const float*)d_in[6],  (const float*)d_in[9],
                        (const float*)d_in[12], (const float*)d_in[15]};
  const float* Wh[4] = {(const float*)d_in[7],  (const float*)d_in[10],
                        (const float*)d_in[13], (const float*)d_in[16]};
  const float* bb[4] = {(const float*)d_in[8],  (const float*)d_in[11],
                        (const float*)d_in[14], (const float*)d_in[17]};
  const float* mlp_w = (const float*)d_in[18];
  const float* mlp_b = (const float*)d_in[19];
  const float* ada_w = (const float*)d_in[20];
  const float* ada_b = (const float*)d_in[21];

  float* out           = (float*)d_out;
  float* windows_y     = out;              // (1024,64,24)
  float* windows_y_hat = out + 1572864;    // (1024,64,24)
  float* levels        = out + 3145728;    // (64,1071)

  char* ws = (char*)d_ws;
  size_t off = 0;
  auto alloc = [&](size_t bytes) -> void* {
    void* p = ws + off;
    off = (off + bytes + 255) & ~(size_t)255;
    return p;
  };
  unsigned*       counters = (unsigned*)alloc(64 * 4);
  float*          seas     = (float*)alloc((size_t)64 * 1095 * 4);
  unsigned short* X0       = (unsigned short*)alloc((size_t)65536 * 32 * 2);
  int KCtot[4] = {17, 32, 32, 32};                       // [Wh(16)|Wx(1 or 16)]
  unsigned short* Wp[4];
  for (int i = 0; i < 4; ++i)
    Wp[i] = (unsigned short*)alloc((size_t)KCtot[i] * 128 * 512 * 2);
  unsigned short* WpM  = (unsigned short*)alloc((size_t)16 * 32 * 512 * 2);
  unsigned short* WpA  = (unsigned short*)alloc((size_t)16 * 2 * 512 * 2);
  float*          Cst  = (float*)alloc((size_t)512 * 512 * 4);
  unsigned short* seqA = (unsigned short*)alloc((size_t)65536 * 512 * 2);
  unsigned short* seqB = (unsigned short*)alloc((size_t)65536 * 512 * 2);
  unsigned short* seqC = (unsigned short*)alloc((size_t)65536 * 512 * 2);

  init_counters_kernel<<<1, 64, 0, stream>>>(counters);
  es_forward_kernel<<<1, 64, 0, stream>>>(y, idxs, lev_sms, seas_sms, init_seas,
                                          levels, seas);
  build_windows_kernel<<<256, 256, 0, stream>>>(y, categories, levels, seas,
                                                X0, windows_y);

  int Din[4] = {30, 512, 512, 512};
  for (int i = 0; i < 4; ++i) {
    int total = KCtot[i] * 128 * 512;
    pack_b_kernel<<<(total + 255) / 256, 256, 0, stream>>>(
        Wh[i], 512, Wx[i], Din[i], 2048, Wp[i], 128, total);
  }
  {
    int total = 16 * 32 * 512;
    pack_b_kernel<<<(total + 255) / 256, 256, 0, stream>>>(
        mlp_w, 512, (const float*)nullptr, 0, 512, WpM, 32, total);
  }
  {
    int total = 16 * 2 * 512;
    pack_b_kernel<<<(total + 255) / 256, 256, 0, stream>>>(
        ada_w, 512, (const float*)nullptr, 0, 24, WpA, 2, total);
  }

  const int NBLK = 32;   // 256 waves; co-resident for the grid barrier
  // group 0: dilation 1 then 2
  lstm_scan_kernel<<<NBLK, 256, 0, stream>>>(X0,   32,  1, seqA, Wp[0], bb[0],
                                             Cst, counters + 0, 1024,  64, NBLK);
  lstm_scan_kernel<<<NBLK, 256, 0, stream>>>(seqA, 512, 16, seqB, Wp[1], bb[1],
                                             Cst, counters + 1,  512, 128, NBLK);
  // group 1: dilation 4 then 8, then residual vs group-0 output (seqB)
  lstm_scan_kernel<<<NBLK, 256, 0, stream>>>(seqB, 512, 16, seqA, Wp[2], bb[2],
                                             Cst, counters + 2,  256, 256, NBLK);
  lstm_scan_kernel<<<NBLK, 256, 0, stream>>>(seqA, 512, 16, seqC, Wp[3], bb[3],
                                             Cst, counters + 3,  128, 512, NBLK);
  size_t nres = (size_t)65536 * 512;
  residual_add_kernel<<<(unsigned)((nres + 255) / 256), 256, 0, stream>>>(seqC, seqB, nres);

  // MLP: tanh(seqC @ mlp_w + mlp_b) -> seqA (bf16), 4-wide N blocking
  gemm_wmma_kernel<<<1024, 256, 0, stream>>>(seqC, 512, WpM, mlp_b, 512,
                                             4096, 32, 16, seqA, 512,
                                             (float*)nullptr, 0, 1);
  // Adapter: seqA @ ada_w + ada_b -> windows_y_hat (f32, N=24 of padded 32)
  gemm_wmma_kernel<<<256, 256, 0, stream>>>(seqA, 512, WpA, ada_b, 24,
                                            4096, 2, 16,
                                            (unsigned short*)nullptr, 0,
                                            windows_y_hat, 24, 0);
}
